// FAM_82463372083966
// MI455X (gfx1250) — matrix operand order repaired and checked by reading, not compiled, not run on hardware
//
#include <hip/hip_runtime.h>
#include <hip/hip_bf16.h>
#include <math.h>
#include <stdint.h>

#define DEV static __device__ __forceinline__

typedef __bf16 bf16;
typedef __attribute__((ext_vector_type(16))) __bf16 v16bf;
typedef __attribute__((ext_vector_type(8)))  __bf16 v8bf;
typedef __attribute__((ext_vector_type(8)))  float  v8f;

// Async-copy builtin wants int4-element pointers (gcc vector_size style),
// global (AS1) source and LDS (AS3) destination.
typedef int v4i_vs __attribute__((vector_size(16)));
typedef __attribute__((address_space(1))) v4i_vs gv4i;
typedef __attribute__((address_space(3))) v4i_vs lv4i;

static constexpr int Cc = 256;
static constexpr int Nn = 4096;
static constexpr int CR = 32;
static constexpr int KC = 64;   // K-chunk staged in LDS for the big GEMM

#if __has_builtin(__builtin_amdgcn_global_load_async_to_lds_b128) && \
    __has_builtin(__builtin_amdgcn_s_wait_asynccnt)
#define HAS_ASYNC 1
#else
#define HAS_ASYNC 0
#endif

DEV bf16 tobf(float f) { return (bf16)f; }

DEV v8f zero8() {
  v8f z;
#pragma unroll
  for (int i = 0; i < 8; ++i) z[i] = 0.f;
  return z;
}

template <int NCNT> DEV void wait_async() {
#if HAS_ASYNC
  __builtin_amdgcn_s_wait_asynccnt(NCNT);
#endif
}

// A fragment: A is [16 x K] row-major (ld = K-stride), tile starts at base, k offset k0.
// ISA layout: lanes 0-15 -> M=lane, halves {K=kb..kb+7, K=16+kb..16+kb+7}, kb=(lane>=16)?8:0
DEV v16bf load_fragA(const bf16* __restrict__ base, int ld, int k0) {
  const int lane = threadIdx.x & 31;
  const int row  = lane & 15;
  const int kb   = (lane >> 4) << 3;  // 0 or 8
  const bf16* p = base + row * ld + k0 + kb;
  v8bf lo = *reinterpret_cast<const v8bf*>(p);
  v8bf hi = *reinterpret_cast<const v8bf*>(p + 16);
  return __builtin_shufflevector(lo, hi, 0,1,2,3,4,5,6,7,8,9,10,11,12,13,14,15);
}

// B fragment: logical B is [K x 16]; we store B^T row-major as [16 x K] (ld = K-stride).
// ISA layout: lanes 0-15 -> N=lane, halves {K=kb..kb+7, K=16+kb..16+kb+7}, kb=(lane>=16)?8:0
DEV v16bf load_fragB(const bf16* __restrict__ baseT, int ld, int k0) {
  const int lane = threadIdx.x & 31;
  const int col  = lane & 15;
  const int kb   = (lane >> 4) << 3;  // 0 or 8
  const bf16* p = baseT + col * ld + k0 + kb;
  v8bf lo = *reinterpret_cast<const v8bf*>(p);
  v8bf hi = *reinterpret_cast<const v8bf*>(p + 16);
  return __builtin_shufflevector(lo, hi, 0,1,2,3,4,5,6,7,8,9,10,11,12,13,14,15);
}

DEV v8f wmma_bf16(v16bf a, v16bf b, v8f c) {
  return __builtin_amdgcn_wmma_f32_16x16x32_bf16(false, a, false, b, (short)0, c, false, false);
}

// Copy `rows` x KC bf16 tile (row stride ld, k offset k0) from global into LDS.
// Uses CDNA5 async global->LDS DMA (ASYNCcnt) when available: 16B per lane-op.
DEV void stage_rows(const bf16* __restrict__ g, int ld, int k0,
                    bf16* __restrict__ l, int rows) {
  const int lane = threadIdx.x & 31;
  const int nchunk = rows * (KC / 8);       // 16B chunks
  for (int q = lane; q < nchunk; q += 32) {
    const int row = q >> 3;
    const int ko  = (q & 7) * 8;
    const bf16* gp = g + (size_t)row * ld + k0 + ko;
    bf16* lp = l + row * KC + ko;
#if HAS_ASYNC
    __builtin_amdgcn_global_load_async_to_lds_b128(
        (gv4i*)(uintptr_t)gp,
        (lv4i*)(uint32_t)(uintptr_t)lp,
        0, 0);
#else
    *reinterpret_cast<v8bf*>(lp) = *reinterpret_cast<const v8bf*>(gp);
#endif
  }
}

// ---------------- kernels ----------------

__global__ void k_cvt_w(const float* __restrict__ vw, const float* __restrict__ tw,
                        bf16* __restrict__ vwb, bf16* __restrict__ twb, int n) {
  int i = blockIdx.x * blockDim.x + threadIdx.x;
  if (i < n) { vwb[i] = tobf(vw[i]); twb[i] = tobf(tw[i]); }
}

// qk = qk_w @ x  (one thread per column n), then bn1/bn2+relu -> q[N,32], kT[N,32], row/col sums
__global__ void k_proj_qk(const float* __restrict__ x, const float* __restrict__ qkw,
                          const float* __restrict__ s1, const float* __restrict__ b1,
                          const float* __restrict__ s2, const float* __restrict__ b2,
                          bf16* __restrict__ qb, bf16* __restrict__ ktb,
                          float* __restrict__ qsum, float* __restrict__ ksum) {
  int n = blockIdx.x * blockDim.x + threadIdx.x;
  if (n >= Nn) return;
  float acc[CR];
#pragma unroll
  for (int o = 0; o < CR; ++o) acc[o] = 0.f;
  for (int c = 0; c < Cc; ++c) {
    float xv = x[(size_t)c * Nn + n];
#pragma unroll
    for (int o = 0; o < CR; ++o) acc[o] += qkw[o * Cc + c] * xv;
  }
  float qs = 0.f, ks = 0.f;
#pragma unroll
  for (int o = 0; o < CR; ++o) {
    float t = acc[o];
    float q = fmaxf(t * s1[o] + b1[o], 0.f);
    float k = fmaxf(t * s2[o] + b2[o], 0.f);
    qb[(size_t)n * CR + o]  = tobf(q);
    ktb[(size_t)n * CR + o] = tobf(k);
    qs += q; ks += k;
  }
  qsum[n] = qs; ksum[n] = ks;
}

// x^T in bf16: xtb[n*C + c] = x[c*N + n]
__global__ void k_tr_x(const float* __restrict__ x, bf16* __restrict__ xtb) {
  int i = blockIdx.x * blockDim.x + threadIdx.x;  // [0, C*N)
  int c = i & (Cc - 1);
  int n = i >> 8;
  xtb[i] = tobf(x[(size_t)c * Nn + n]);
}

// v = relu(bn3(v_w @ x + v_b)) via WMMA; vb is [C, N] bf16 row-major
__global__ void k_proj_v(const bf16* __restrict__ vwb, const bf16* __restrict__ xtb,
                         const float* __restrict__ vbias,
                         const float* __restrict__ s3, const float* __restrict__ b3,
                         bf16* __restrict__ vb) {
  const int n0 = blockIdx.x * 16;
  const int o0 = blockIdx.y * 16;
  v8f acc = zero8();
  for (int k0 = 0; k0 < Cc; k0 += 32) {
    v16bf a = load_fragA(vwb + o0 * Cc, Cc, k0);
    v16bf b = load_fragB(xtb + (size_t)n0 * Cc, Cc, k0);
    acc = wmma_bf16(a, b, acc);
  }
  const int lane = threadIdx.x & 31;
  const int col = lane & 15, hi = lane >> 4;
  const int n = n0 + col;
#pragma unroll
  for (int r = 0; r < 8; ++r) {
    int o = o0 + r + 8 * hi;
    float t = acc[r] + vbias[o];
    t = fmaxf(t * s3[o] + b3[o], 0.f);
    vb[(size_t)o * Nn + n] = tobf(t);
  }
}

// Pass 1: per-row softmax stats over energy = qK - qsum*ksum (recomputed tiles, online)
__global__ void k_rowstats(const bf16* __restrict__ qb, const bf16* __restrict__ ktb,
                           const float* __restrict__ qsum, const float* __restrict__ ksum,
                           float* __restrict__ rowmax, float* __restrict__ rowinv) {
  const int n0 = blockIdx.x * 16;
  const int lane = threadIdx.x & 31;
  const int col = lane & 15, hi = lane >> 4;
  v16bf a = load_fragA(qb + (size_t)n0 * CR, CR, 0);
  float qs[8], m[8], s[8];
#pragma unroll
  for (int r = 0; r < 8; ++r) {
    qs[r] = qsum[n0 + r + 8 * hi];
    m[r] = -3.0e38f; s[r] = 0.f;
  }
  for (int mt = 0; mt < Nn; mt += 16) {
    v16bf b = load_fragB(ktb + (size_t)mt * CR, CR, 0);
    v8f c = zero8();
    c = wmma_bf16(a, b, c);
    float kv = ksum[mt + col];
#pragma unroll
    for (int r = 0; r < 8; ++r) {
      float e  = c[r] - qs[r] * kv;
      float nm = fmaxf(m[r], e);
      s[r] = s[r] * __expf(m[r] - nm) + __expf(e - nm);
      m[r] = nm;
    }
  }
  // reduce (max,sum) across the 16 lanes holding the same rows
#pragma unroll
  for (int d = 1; d < 16; d <<= 1) {
#pragma unroll
    for (int r = 0; r < 8; ++r) {
      float om = __shfl_xor(m[r], d, 32);
      float os = __shfl_xor(s[r], d, 32);
      float nm = fmaxf(m[r], om);
      s[r] = s[r] * __expf(m[r] - nm) + os * __expf(om - nm);
      m[r] = nm;
    }
  }
  if (col == 0) {
#pragma unroll
    for (int r = 0; r < 8; ++r) {
      int n = n0 + r + 8 * hi;
      rowmax[n] = m[r];
      rowinv[n] = 1.0f / s[r];
    }
  }
}

// Pass 2: att = exp(e - rowmax)*rowinv, stored transposed bf16: attT[m*N + n]
__global__ void k_att(const bf16* __restrict__ qb, const bf16* __restrict__ ktb,
                      const float* __restrict__ qsum, const float* __restrict__ ksum,
                      const float* __restrict__ rowmax, const float* __restrict__ rowinv,
                      bf16* __restrict__ attT) {
  const int n0 = blockIdx.x * 16;
  const int lane = threadIdx.x & 31;
  const int col = lane & 15, hi = lane >> 4;
  v16bf a = load_fragA(qb + (size_t)n0 * CR, CR, 0);
  float qs[8], rm[8], ri[8];
#pragma unroll
  for (int r = 0; r < 8; ++r) {
    int n = n0 + r + 8 * hi;
    qs[r] = qsum[n]; rm[r] = rowmax[n]; ri[r] = rowinv[n];
  }
  for (int mt = 0; mt < Nn; mt += 16) {
    v16bf b = load_fragB(ktb + (size_t)mt * CR, CR, 0);
    v8f c = zero8();
    c = wmma_bf16(a, b, c);
    float kv = ksum[mt + col];
    v8bf outv;
#pragma unroll
    for (int r = 0; r < 8; ++r) {
      float e = c[r] - qs[r] * kv;
      outv[r] = tobf(__expf(e - rm[r]) * ri[r]);
    }
    *reinterpret_cast<v8bf*>(attT + (size_t)(mt + col) * Nn + n0 + 8 * hi) = outv;
  }
}

// colinv[m] = 1 / (1e-9 + sum_n att[n][m])   (attT rows are contiguous)
__global__ void k_colsum(const bf16* __restrict__ attT, float* __restrict__ colinv) {
  int m = blockIdx.x * blockDim.x + threadIdx.x;
  if (m >= Nn) return;
  const bf16* p = attT + (size_t)m * Nn;
  float s = 0.f;
  for (int n = 0; n < Nn; n += 8) {
    v8bf v = *reinterpret_cast<const v8bf*>(p + n);
#pragma unroll
    for (int j = 0; j < 8; ++j) s += (float)v[j];
  }
  colinv[m] = 1.0f / (1e-9f + s);
}

// x_r = (v @ att) * colinv[m]; diffT[m*C + c] = bf16(x[c][m] - x_r[c][m]).
// Big GEMM (K=4096): async global->LDS double-buffered staging, 2x C-blocking per wave.
__global__ void k_xr_diff(const bf16* __restrict__ vb, const bf16* __restrict__ attT,
                          const float* __restrict__ colinv, const float* __restrict__ x,
                          bf16* __restrict__ diffT) {
  __shared__ __align__(16) bf16 lA[2][32 * KC];
  __shared__ __align__(16) bf16 lB[2][16 * KC];
  const int m0 = blockIdx.x * 16;
  const int c0 = blockIdx.y * 32;
  const int lane = threadIdx.x & 31;
  const int col = lane & 15, hi = lane >> 4;
  const bf16* arow = vb + (size_t)c0 * Nn;
  const bf16* brow = attT + (size_t)m0 * Nn;

  v8f acc0 = zero8(), acc1 = zero8();

  // prologue: stage chunk 0 (12 async b128 per lane: 8 for A(32xKC), 4 for B(16xKC))
  stage_rows(arow, Nn, 0, lA[0], 32);
  stage_rows(brow, Nn, 0, lB[0], 16);

  int buf = 0;
  for (int k0 = 0; k0 < Nn; k0 += KC) {
    const int nk = k0 + KC;
    if (nk < Nn) {
      stage_rows(arow, Nn, nk, lA[buf ^ 1], 32);
      stage_rows(brow, Nn, nk, lB[buf ^ 1], 16);
      wait_async<12>();      // current chunk's 12 ops done; next chunk in flight
    } else {
      wait_async<0>();
    }
#pragma unroll
    for (int ks = 0; ks < KC; ks += 32) {
      v16bf bfr = load_fragB(lB[buf], KC, ks);
      v16bf a0  = load_fragA(lA[buf], KC, ks);
      v16bf a1  = load_fragA(lA[buf] + 16 * KC, KC, ks);
      acc0 = wmma_bf16(a0, bfr, acc0);
      acc1 = wmma_bf16(a1, bfr, acc1);
    }
    buf ^= 1;
  }

  const int m = m0 + col;
  const float ci = colinv[m];
#pragma unroll
  for (int ct = 0; ct < 2; ++ct) {
    v8f& acc = ct ? acc1 : acc0;
    v8bf dv;
#pragma unroll
    for (int r = 0; r < 8; ++r) {
      int c = c0 + ct * 16 + 8 * hi + r;
      dv[r] = tobf(x[(size_t)c * Nn + m] - acc[r] * ci);
    }
    *reinterpret_cast<v8bf*>(diffT + (size_t)m * Cc + c0 + ct * 16 + 8 * hi) = dv;
  }
}

// out = x + relu(an(trans_w @ diff + trans_b))
__global__ void k_trans_out(const bf16* __restrict__ twb, const bf16* __restrict__ diffT,
                            const float* __restrict__ tb,
                            const float* __restrict__ ans, const float* __restrict__ anb,
                            const float* __restrict__ x, float* __restrict__ out) {
  const int n0 = blockIdx.x * 16;
  const int o0 = blockIdx.y * 16;
  v8f acc = zero8();
  for (int k0 = 0; k0 < Cc; k0 += 32) {
    v16bf a = load_fragA(twb + o0 * Cc, Cc, k0);
    v16bf b = load_fragB(diffT + (size_t)n0 * Cc, Cc, k0);
    acc = wmma_bf16(a, b, acc);
  }
  const int lane = threadIdx.x & 31;
  const int col = lane & 15, hi = lane >> 4;
  const int n = n0 + col;
#pragma unroll
  for (int r = 0; r < 8; ++r) {
    int o = o0 + r + 8 * hi;
    float t = acc[r] + tb[o];
    t = fmaxf(t * ans[o] + anb[o], 0.f);
    out[(size_t)o * Nn + n] = x[(size_t)o * Nn + n] + t;
  }
}

// ---------------- host ----------------

extern "C" void kernel_launch(void* const* d_in, const int* in_sizes, int n_in,
                              void* d_out, int out_size, void* d_ws, size_t ws_size,
                              hipStream_t stream) {
  const float* x       = (const float*)d_in[0];
  const float* qk_w    = (const float*)d_in[1];
  const float* v_w     = (const float*)d_in[2];
  const float* v_b     = (const float*)d_in[3];
  const float* trans_w = (const float*)d_in[4];
  const float* trans_b = (const float*)d_in[5];
  const float* bn1_s   = (const float*)d_in[6];
  const float* bn1_b   = (const float*)d_in[7];
  const float* bn2_s   = (const float*)d_in[8];
  const float* bn2_b   = (const float*)d_in[9];
  const float* bn3_s   = (const float*)d_in[10];
  const float* bn3_b   = (const float*)d_in[11];
  const float* an_s    = (const float*)d_in[12];
  const float* an_b    = (const float*)d_in[13];
  float* out = (float*)d_out;

  char* ws = (char*)d_ws;
  size_t off = 0;
  auto take = [&](size_t bytes) -> char* {
    char* p = ws + off;
    off += (bytes + 255) & ~(size_t)255;
    return p;
  };
  bf16*  vwb    = (bf16*)take((size_t)Cc * Cc * 2);
  bf16*  twb    = (bf16*)take((size_t)Cc * Cc * 2);
  bf16*  qb     = (bf16*)take((size_t)Nn * CR * 2);
  bf16*  ktb    = (bf16*)take((size_t)Nn * CR * 2);
  float* qsum   = (float*)take((size_t)Nn * 4);
  float* ksum   = (float*)take((size_t)Nn * 4);
  bf16*  xtb    = (bf16*)take((size_t)Nn * Cc * 2);
  bf16*  vb     = (bf16*)take((size_t)Cc * Nn * 2);
  float* rowmax = (float*)take((size_t)Nn * 4);
  float* rowinv = (float*)take((size_t)Nn * 4);
  bf16*  attT   = (bf16*)take((size_t)Nn * Nn * 2);
  float* colinv = (float*)take((size_t)Nn * 4);
  bf16*  diffT  = (bf16*)take((size_t)Nn * Cc * 2);

  k_cvt_w<<<(Cc * Cc + 255) / 256, 256, 0, stream>>>(v_w, trans_w, vwb, twb, Cc * Cc);

  for (int b = 0; b < 4; ++b) {
    const float* xb = x   + (size_t)b * Cc * Nn;
    float*       ob = out + (size_t)b * Cc * Nn;

    k_proj_qk<<<Nn / 256, 256, 0, stream>>>(xb, qk_w, bn1_s, bn1_b, bn2_s, bn2_b,
                                            qb, ktb, qsum, ksum);
    k_tr_x<<<(Cc * Nn) / 256, 256, 0, stream>>>(xb, xtb);
    k_proj_v<<<dim3(Nn / 16, Cc / 16), 32, 0, stream>>>(vwb, xtb, v_b, bn3_s, bn3_b, vb);
    k_rowstats<<<Nn / 16, 32, 0, stream>>>(qb, ktb, qsum, ksum, rowmax, rowinv);
    k_att<<<Nn / 16, 32, 0, stream>>>(qb, ktb, qsum, ksum, rowmax, rowinv, attT);
    k_colsum<<<Nn / 256, 256, 0, stream>>>(attT, colinv);
    k_xr_diff<<<dim3(Nn / 16, Cc / 32), 32, 0, stream>>>(vb, attT, colinv, xb, diffT);
    k_trans_out<<<dim3(Nn / 16, Cc / 16), 32, 0, stream>>>(twb, diffT, trans_b, an_s, an_b,
                                                           xb, ob);
  }
}
